// HeteroGCN_49976239456893
// MI455X (gfx1250) — matrix operand order repaired and checked by reading, not compiled, not run on hardware
//
#include <hip/hip_runtime.h>

#define NU 100000
#define NI 50000
#define CH 128
#define NE 600000
#define LEPS 1e-5f

// LDS pitches chosen for bank-conflict-free WMMA fragment loads:
//  - bf16 arrays: pitch 136 elems = 272 B = 68 dwords -> lane n hits bank 4n mod 64
//  - f32 D tile : pitch 132 elems -> lane halves land on disjoint bank ranges
#define WP 136
#define DP 132

typedef __attribute__((ext_vector_type(16))) __bf16 v16bf;
typedef __attribute__((ext_vector_type(8)))  __bf16 v8bf;
typedef __attribute__((ext_vector_type(8)))  float  v8f;

__device__ __forceinline__ v16bf cat16(v8bf lo, v8bf hi) {
    return __builtin_shufflevector(lo, hi, 0,1,2,3,4,5,6,7,8,9,10,11,12,13,14,15);
}

// ---------------------------------------------------------------------------
// Scatter-mean accumulation: 32 lanes per edge, float4 gather, f32 atomics.
// agg[dst] += x_src[src]; cnt[dst] += 1
// ---------------------------------------------------------------------------
__global__ __launch_bounds__(256) void scatter_accum(
    const float* __restrict__ xsrc,
    const int*   __restrict__ src,
    const int*   __restrict__ dst,
    float*       __restrict__ agg,
    float*       __restrict__ cnt)
{
    int gtid = blockIdx.x * 256 + threadIdx.x;
    int e    = gtid >> 5;
    int lane = gtid & 31;
    if (e >= NE) return;
    int s = src[e];
    int d = dst[e];
    float4 v = ((const float4*)(xsrc + (size_t)s * CH))[lane];
    float* o = agg + (size_t)d * CH + lane * 4;
    atomicAdd(o + 0, v.x);
    atomicAdd(o + 1, v.y);
    atomicAdd(o + 2, v.z);
    atomicAdd(o + 3, v.w);
    if (lane == 0) atomicAdd(cnt + d, 1.0f);
}

// ---------------------------------------------------------------------------
// Fused: D = (agg/max(cnt,1)) @ Wrel^T + x @ Wroot^T + brel ; LayerNorm ; ReLU
// Block = 16 rows x 128 cols. 8 wave32 waves, each owns a 16x16 N-tile.
// bf16 WMMA (v_wmma_f32_16x16x32_bf16), f32 accumulate.
// ---------------------------------------------------------------------------
__global__ __launch_bounds__(256) void gemm_ln_relu(
    const float* __restrict__ agg, const float* __restrict__ cnt,
    const float* __restrict__ x,
    const float* __restrict__ Wrel, const float* __restrict__ brel,
    const float* __restrict__ Wroot,
    const float* __restrict__ gamma, const float* __restrict__ beta,
    float* __restrict__ out)
{
    __shared__ __attribute__((aligned(16))) __bf16 Wrel_s [CH * WP]; // ~34 KB
    __shared__ __attribute__((aligned(16))) __bf16 Wroot_s[CH * WP]; // ~34 KB
    __shared__ __attribute__((aligned(16))) __bf16 Amean_s[16 * WP]; // ~4.3 KB
    __shared__ __attribute__((aligned(16))) __bf16 Ax_s   [16 * WP]; // ~4.3 KB
    __shared__ __attribute__((aligned(16))) float  D_s    [16 * DP]; // ~8.4 KB

    const int tid = threadIdx.x;
    const int r0  = blockIdx.x * 16;

    // Stage weights f32->bf16 into padded LDS rows.
    // Thread t: row = t/2, cols [64*(t&1), 64*(t&1)+63], float4 loads.
    {
        const int wrow = tid >> 1;
        const int cbase = (tid & 1) * 64;
        const float4* ga = (const float4*)(Wrel  + wrow * CH + cbase);
        const float4* gb = (const float4*)(Wroot + wrow * CH + cbase);
        __bf16* la = Wrel_s  + wrow * WP + cbase;
        __bf16* lb = Wroot_s + wrow * WP + cbase;
        #pragma unroll
        for (int i = 0; i < 16; ++i) {
            float4 a = ga[i];
            float4 b = gb[i];
            int o = i * 4;
            la[o + 0] = (__bf16)a.x; la[o + 1] = (__bf16)a.y;
            la[o + 2] = (__bf16)a.z; la[o + 3] = (__bf16)a.w;
            lb[o + 0] = (__bf16)b.x; lb[o + 1] = (__bf16)b.y;
            lb[o + 2] = (__bf16)b.z; lb[o + 3] = (__bf16)b.w;
        }
    }
    // Stage A tiles (mean and x rows) f32->bf16: row = tid/16, 8 cols each.
    {
        const int row = tid >> 4;
        const int c0  = (tid & 15) * 8;
        const size_t g = (size_t)(r0 + row) * CH + c0;
        const float inv = 1.0f / fmaxf(cnt[r0 + row], 1.0f);
        #pragma unroll
        for (int i = 0; i < 8; ++i) {
            Amean_s[row * WP + c0 + i] = (__bf16)(agg[g + i] * inv);
            Ax_s   [row * WP + c0 + i] = (__bf16)(x  [g + i]);
        }
    }
    __syncthreads();

    const int wave = tid >> 5;
    const int lane = tid & 31;
    const int n0   = wave * 16;        // this wave's N-tile origin
    const int n    = lane & 15;        // A row / B,C column within tile
    const int koff = (lane < 16) ? 0 : 8;   // A-matrix K sublane offset
    const int boff = (lane < 16) ? 0 : 16;  // unused: B loads use koff, like A

    // Fold bias into accumulator init (same column for both lane halves).
    const float bv = brel[n0 + n];
    v8f c = { bv, bv, bv, bv, bv, bv, bv, bv };

    #pragma unroll
    for (int kb = 0; kb < CH; kb += 32) {
        // A fragment (16x32 bf16): lane holds K {koff..koff+7} U {16+koff..}
        v16bf amean = cat16(*(const v8bf*)&Amean_s[n * WP + kb + koff],
                            *(const v8bf*)&Amean_s[n * WP + kb + 16 + koff]);
        v16bf ax    = cat16(*(const v8bf*)&Ax_s   [n * WP + kb + koff],
                            *(const v8bf*)&Ax_s   [n * WP + kb + 16 + koff]);
        // B fragment (32x16 bf16): B[k][n] = W[n][k]; lane holds K
        // {koff..koff+7} U {16+koff..} of W row (n0+n), same as A.
        v16bf brl = cat16(*(const v8bf*)&Wrel_s [(n0 + n) * WP + kb + koff],
                          *(const v8bf*)&Wrel_s [(n0 + n) * WP + kb + 16 + koff]);
        v16bf brt = cat16(*(const v8bf*)&Wroot_s[(n0 + n) * WP + kb + koff],
                          *(const v8bf*)&Wroot_s[(n0 + n) * WP + kb + 16 + koff]);

        c = __builtin_amdgcn_wmma_f32_16x16x32_bf16(
                false, amean, false, brl, (short)0, c, false, false);
        c = __builtin_amdgcn_wmma_f32_16x16x32_bf16(
                false, ax,    false, brt, (short)0, c, false, false);
    }

    // Scatter accumulator tile to LDS per ISA C/D layout:
    // lanes 0-15: rows 0..7 ; lanes 16-31: rows 8..15 (same column n).
    {
        const int rbase = (lane < 16) ? 0 : 8;
        #pragma unroll
        for (int r = 0; r < 8; ++r)
            D_s[(rbase + r) * DP + n0 + n] = c[r];
    }
    __syncthreads();

    // LayerNorm + ReLU: 16 threads per row, 8 cols each, shfl_xor reduce.
    {
        const int row = tid >> 4;
        const int c0  = (tid & 15) * 8;
        float s = 0.0f, s2 = 0.0f;
        float vals[8];
        #pragma unroll
        for (int i = 0; i < 8; ++i) {
            float v = D_s[row * DP + c0 + i];
            vals[i] = v; s += v; s2 += v * v;
        }
        #pragma unroll
        for (int m = 8; m >= 1; m >>= 1) {
            s  += __shfl_xor(s,  m, 16);
            s2 += __shfl_xor(s2, m, 16);
        }
        const float mu   = s * (1.0f / CH);
        const float var  = s2 * (1.0f / CH) - mu * mu;
        const float rstd = rsqrtf(var + LEPS);
        #pragma unroll
        for (int i = 0; i < 8; ++i) {
            const int j = c0 + i;
            float v = (vals[i] - mu) * rstd * gamma[j] + beta[j];
            out[(size_t)(r0 + row) * CH + j] = fmaxf(v, 0.0f);
        }
    }
}

// ---------------------------------------------------------------------------
extern "C" void kernel_launch(void* const* d_in, const int* in_sizes, int n_in,
                              void* d_out, int out_size, void* d_ws, size_t ws_size,
                              hipStream_t stream)
{
    const float* x_user = (const float*)d_in[0];
    const float* x_item = (const float*)d_in[1];
    const float* Wrel   = (const float*)d_in[2];  // [L,2,C,C]
    const float* brel   = (const float*)d_in[3];  // [L,2,C]
    const float* Wroot  = (const float*)d_in[4];  // [L,2,C,C]
    const float* gamma  = (const float*)d_in[5];  // [L,2,C] (node type)
    const float* beta   = (const float*)d_in[6];
    const int* src_u2i  = (const int*)d_in[7];
    const int* dst_u2i  = (const int*)d_in[8];
    const int* src_i2u  = (const int*)d_in[9];
    const int* dst_i2u  = (const int*)d_in[10];

    float* out_user = (float*)d_out;
    float* out_item = out_user + (size_t)NU * CH;

    float* agg_u = (float*)d_ws;                   // NU*CH
    float* agg_i = agg_u + (size_t)NU * CH;        // NI*CH
    float* cnt_u = agg_i + (size_t)NI * CH;        // NU
    float* cnt_i = cnt_u + NU;                     // NI

    const int scat_blocks = (NE * 32) / 256;       // 75000

    for (int l = 0; l < 2; ++l) {
        const float* xu = (l == 0) ? x_user : out_user;
        const float* xi = (l == 0) ? x_item : out_item;

        hipMemsetAsync(agg_u, 0, (size_t)NU * CH * sizeof(float), stream);
        hipMemsetAsync(agg_i, 0, (size_t)NI * CH * sizeof(float), stream);
        hipMemsetAsync(cnt_u, 0, (size_t)NU * sizeof(float), stream);
        hipMemsetAsync(cnt_i, 0, (size_t)NI * sizeof(float), stream);

        // edge type 0: user->item ; edge type 1: item->user
        scatter_accum<<<scat_blocks, 256, 0, stream>>>(xu, src_u2i, dst_u2i, agg_i, cnt_i);
        scatter_accum<<<scat_blocks, 256, 0, stream>>>(xi, src_i2u, dst_i2u, agg_u, cnt_u);

        // items: relation (l,0), node-type 1 for LN params
        gemm_ln_relu<<<NI / 16, 256, 0, stream>>>(
            agg_i, cnt_i, xi,
            Wrel  + ((size_t)l * 2 + 0) * CH * CH,
            brel  + ((size_t)l * 2 + 0) * CH,
            Wroot + ((size_t)l * 2 + 0) * CH * CH,
            gamma + ((size_t)l * 2 + 1) * CH,
            beta  + ((size_t)l * 2 + 1) * CH,
            out_item);

        // users: relation (l,1), node-type 0 for LN params
        gemm_ln_relu<<<NU / 16, 256, 0, stream>>>(
            agg_u, cnt_u, xu,
            Wrel  + ((size_t)l * 2 + 1) * CH * CH,
            brel  + ((size_t)l * 2 + 1) * CH,
            Wroot + ((size_t)l * 2 + 1) * CH * CH,
            gamma + ((size_t)l * 2 + 0) * CH,
            beta  + ((size_t)l * 2 + 0) * CH,
            out_user);
    }
}
